// nodesFC_BILSTM_24275155157316
// MI455X (gfx1250) — compile-verified, matrix operand order checked by reading, not run on hardware
//
#include <hip/hip_runtime.h>
#include <hip/hip_bf16.h>
#include <math.h>

// ---------------- constants ----------------
#define BB 512          // batch
#define CC 184          // cities (sequence length of LSTM scan)
#define IND 14          // in_dim
#define DD 15           // in_dim+1 (LSTM layer0 input width)
#define HH 32           // hidden
#define G4 128          // 4*H gates
#define PRED 24
#define HIST 24
#define KMLP 2576       // C*IND
#define KMLPP 2592      // padded to 81*32
#define NMLPP 192       // 184 padded to 12*16
#define RT_TOT 5888     // C*B/16 row tiles

typedef __attribute__((ext_vector_type(16))) __bf16 v16bf;
typedef __attribute__((ext_vector_type(8)))  __bf16 v8bf;
typedef __attribute__((ext_vector_type(8)))  float  v8f;

static __device__ __forceinline__ __bf16 tobf(float f) { return (__bf16)f; }
static __device__ __forceinline__ float sigf(float x) { return 1.0f / (1.0f + expf(-x)); }

// A/B 16-bit WMMA layout: lane l -> row (l&15), kh = l>>4.
// element j<8  -> k = kh*8 + j
// element j>=8 -> k = 16 + kh*8 + (j-8)
static __device__ __forceinline__ int kf(int kh, int j) {
    return (j < 8) ? (kh * 8 + j) : (16 + kh * 8 + (j - 8));
}

// load 16 bf16 elements for one lane from a contiguous K-run (two 16B chunks)
static __device__ __forceinline__ v16bf load16(const __bf16* p, int kh) {
    v8bf lo = *reinterpret_cast<const v8bf*>(p + kh * 8);
    v8bf hi = *reinterpret_cast<const v8bf*>(p + 16 + kh * 8);
    return __builtin_shufflevector(lo, hi, 0,1,2,3,4,5,6,7,8,9,10,11,12,13,14,15);
}

static __device__ __forceinline__ v8f wmma_bf16(v16bf a, v16bf b, v8f c) {
    return __builtin_amdgcn_wmma_f32_16x16x32_bf16(false, a, false, b, (short)0, c,
                                                   false, false);
}

// ---------------- once-per-call prep ----------------
__global__ void pack_weights_kernel(const float* Wmlp, const float* Wih0,
                                    const float* Whh0, const float* Wih1,
                                    const float* Whh1,
                                    __bf16* WmlpT, __bf16* Wih0b, __bf16* Whh0b,
                                    __bf16* Wih1b, __bf16* Whh1b) {
    long i = (long)blockIdx.x * blockDim.x + threadIdx.x;
    const long nMlp = (long)NMLPP * KMLPP;               // 497664
    if (i < nMlp) {
        int n = (int)(i / KMLPP), k = (int)(i % KMLPP);
        WmlpT[i] = (n < CC && k < KMLP) ? tobf(Wmlp[(long)k * CC + n]) : (__bf16)0.0f;
        return;
    }
    i -= nMlp;
    if (i < 2 * 128 * 32) {                               // Wih0 padded 15->32
        int k = (int)(i & 31); long dw = i >> 5;
        Wih0b[i] = (k < DD) ? tobf(Wih0[dw * DD + k]) : (__bf16)0.0f;
        return;
    }
    i -= 2 * 128 * 32;
    if (i < 2 * 128 * 32) { Whh0b[i] = tobf(Whh0[i]); return; }
    i -= 2 * 128 * 32;
    if (i < 2 * 128 * 64) { Wih1b[i] = tobf(Wih1[i]); return; }
    i -= 2 * 128 * 64;
    if (i < 2 * 128 * 32) { Whh1b[i] = tobf(Whh1[i]); return; }
}

__global__ void init_state_kernel(const float* rain, const float* h0, const float* c0,
                                  float* xn, float* hnP, float* cn) {
    long i = (long)blockIdx.x * blockDim.x + threadIdx.x;
    if (i < (long)BB * CC) {                 // xn = rain_hist[:, -1]
        int b = (int)(i / CC), c = (int)(i % CC);
        xn[i] = rain[((long)b * HIST + (HIST - 1)) * CC + c];
        return;
    }
    i -= (long)BB * CC;
    if (i < 4 * BB * HH) { hnP[i] = h0[i]; return; }
    i -= 4 * BB * HH;
    if (i < 4 * BB * HH) { cn[i] = c0[i]; return; }
}

// ---------------- per-step kernels ----------------
// X14 bf16 [B][2592]: flattened [xn, feat13] per city, K zero-padded
__global__ void pack_x14_kernel(const float* xn, const float* feat, int t, __bf16* X14) {
    long i = (long)blockIdx.x * blockDim.x + threadIdx.x;
    if (i >= (long)BB * KMLPP) return;
    int b = (int)(i / KMLPP), k = (int)(i % KMLPP);
    float v = 0.0f;
    if (k < KMLP) {
        int c = k / IND, j = k % IND;
        v = (j == 0) ? xn[(long)b * CC + c]
                     : feat[(((long)b * (HIST + PRED) + HIST + t) * CC + c) * (IND - 1) + (j - 1)];
    }
    X14[i] = tobf(v);
}

// g = sigmoid(X14 @ WmlpT^T + b); writes x_cat[c][b][0]
__global__ void mlp_gemm_kernel(const __bf16* X14, const __bf16* WT,
                                const float* bmlp, float* xcat) {
    int wg = blockIdx.x * 4 + (threadIdx.x >> 5);   // 0..383
    int lane = threadIdx.x & 31;
    int mt = wg / 12, nt = wg % 12;
    int m15 = lane & 15, kh = lane >> 4;
    const __bf16* arow = X14 + (long)(mt * 16 + m15) * KMLPP;
    const __bf16* brow = WT + (long)(nt * 16 + m15) * KMLPP;
    v8f acc = {};
    for (int kt = 0; kt < KMLPP / 32; ++kt) {
        v16bf a = load16(arow + kt * 32, kh);
        v16bf b = load16(brow + kt * 32, kh);
        acc = wmma_bf16(a, b, acc);
    }
    int c = nt * 16 + m15;
    if (c < CC) {
        float bb = bmlp[c];
        for (int r = 0; r < 8; ++r) {
            int b = mt * 16 + r + 8 * kh;
            xcat[((long)c * BB + b) * DD + 0] = sigf(acc[r] + bb);
        }
    }
}

// x_cat[c][b][1..14] = [xn, feat13]
__global__ void build_xcat_kernel(const float* xn, const float* feat, int t, float* xcat) {
    long i = (long)blockIdx.x * blockDim.x + threadIdx.x;
    if (i >= (long)BB * CC) return;
    int b = (int)(i / CC), c = (int)(i % CC);
    float* dst = xcat + ((long)c * BB + b) * DD;
    dst[1] = xn[(long)b * CC + c];
    const float* f = feat + (((long)b * (HIST + PRED) + HIST + t) * CC + c) * (IND - 1);
    for (int j = 0; j < IND - 1; ++j) dst[2 + j] = f[j];
}

__global__ void mean_u_kernel(const float* xcat, float* u) {
    long i = (long)blockIdx.x * blockDim.x + threadIdx.x;
    if (i >= (long)BB * DD) return;
    int b = (int)(i / DD), d = (int)(i % DD);
    float s = 0.0f;
    for (int c = 0; c < CC; ++c) s += xcat[((long)c * BB + b) * DD + d];
    u[(long)b * 16 + d] = s / (float)CC;
}

__global__ void gru_kernel(const float* u, const float* hin, const float* Wih,
                           const float* Whh, const float* bih, const float* bhh,
                           float* hout) {
    long i = (long)blockIdx.x * blockDim.x + threadIdx.x;
    if (i >= 4L * BB * HH) return;
    int j = (int)(i & 31);
    long sb = i >> 5;                       // s*B+b
    int b = (int)(sb & (BB - 1));
    float rx = bih[j], zx = bih[32 + j], nx = bih[64 + j];
    const float* ub = u + (long)b * 16;
    for (int d = 0; d < DD; ++d) {
        float ud = ub[d];
        rx += ud * Wih[d * 96 + j];
        zx += ud * Wih[d * 96 + 32 + j];
        nx += ud * Wih[d * 96 + 64 + j];
    }
    float rh = bhh[j], zh = bhh[32 + j], nh = bhh[64 + j];
    const float* hrow = hin + sb * HH;
    for (int h = 0; h < HH; ++h) {
        float hv = hrow[h];
        rh += hv * Whh[h * 96 + j];
        zh += hv * Whh[h * 96 + 32 + j];
        nh += hv * Whh[h * 96 + 64 + j];
    }
    float r = sigf(rx + rh), z = sigf(zx + zh);
    float n = tanhf(nx + r * nh);
    hout[i] = (1.0f - z) * n + z * hrow[j];
}

// Hoisted input-side gates: gx[dir][rowtile][nt][lane][8] (f32, WMMA D layout)
__global__ void gates_x_kernel(const float* xcat, const __bf16* yin, int layer,
                               const __bf16* Wih, const float* bias, float* gx) {
    int lane = threadIdx.x & 31;
    int rt = blockIdx.x * 4 + (threadIdx.x >> 5);
    int dir = blockIdx.y;
    int m15 = lane & 15, kh = lane >> 4;
    int row = rt * 16 + m15;
    v16bf a0, a1;
    if (layer == 0) {
        const float* xr = xcat + (long)row * DD;
        for (int j = 0; j < 16; ++j) {
            int k = kf(kh, j);
            a0[j] = (k < DD) ? tobf(xr[k]) : (__bf16)0.0f;
        }
    } else {
        const __bf16* yr = yin + (long)row * 64;
        a0 = load16(yr, kh);
        a1 = load16(yr + 32, kh);
    }
    int din = layer ? 64 : 32;
    for (int nt = 0; nt < 8; ++nt) {
        int n = nt * 16 + m15;
        const __bf16* wr = Wih + (long)(dir * G4 + n) * din;
        v8f acc = {};
        acc = wmma_bf16(a0, load16(wr, kh), acc);
        if (layer) acc = wmma_bf16(a1, load16(wr + 32, kh), acc);
        float bv = bias[dir * G4 + n];
        for (int r = 0; r < 8; ++r) acc[r] += bv;
        float* dst = gx + ((((long)dir * RT_TOT + rt) * 8 + nt) * 32 + lane) * 8;
        *reinterpret_cast<v8f*>(dst) = acc;
    }
}

// Sequential city scan: per city only h @ Whh^T (one WMMA per n-tile) + elementwise.
// One wave per (batch-tile, direction); h relaid A<-D via LDS each step.
// gx tile loads are double-buffered (city c+1 issued before city c's compute) and
// the tile for city c+2 is prefetched into the WGP cache (global_prefetch_b8),
// hiding L2 latency behind the latency-bound recurrence.
__global__ void lstm_scan_kernel(const float* gx, const __bf16* Whh, __bf16* yout,
                                 const float* hin, float* hout, float* cio,
                                 int slicebase) {
    __shared__ __bf16 hbuf[16 * 32];
    int lane = threadIdx.x;
    int bt = blockIdx.x, dir = blockIdx.y;
    int slice = slicebase + dir;
    int b0 = bt * 16;
    int m15 = lane & 15, kh = lane >> 4;

    // per-(dir,city,batch-tile) gx slice base (8KB contiguous per tile)
    const float* gxbase = gx + (((long)dir * RT_TOT + bt) * 8) * 32 * 8 + (long)lane * 8;

    v16bf hA;
    for (int j = 0; j < 16; ++j)
        hA[j] = tobf(hin[((long)slice * BB + b0 + m15) * HH + kf(kh, j)]);

    v8f cacc[2], hcur[2];
    for (int ht = 0; ht < 2; ++ht)
        for (int r = 0; r < 8; ++r) {
            int m = r + 8 * kh, n = ht * 16 + m15;
            cacc[ht][r] = cio[((long)slice * BB + b0 + m) * HH + n];
        }

    v16bf bW[8];
    for (int nt = 0; nt < 8; ++nt)
        bW[nt] = load16(Whh + (long)(dir * G4 + nt * 16 + m15) * HH, kh);

    // double-buffered gx tile loads
    v8f nxt[8];
    {
        int cfirst = dir ? (CC - 1) : 0;
        const float* src = gxbase + (long)cfirst * 32 * (8 * 32 * 8);
        for (int nt = 0; nt < 8; ++nt)
            nxt[nt] = *reinterpret_cast<const v8f*>(src + (long)nt * 32 * 8);
    }

    for (int ci = 0; ci < CC; ++ci) {
        int c = dir ? (CC - 1 - ci) : ci;
        v8f acc[8];
        for (int nt = 0; nt < 8; ++nt) acc[nt] = nxt[nt];

        // issue next city's gx loads + prefetch c+2 before this city's compute
        if (ci + 1 < CC) {
            int cn2 = dir ? (CC - 2 - ci) : (ci + 1);
            const float* src = gxbase + (long)cn2 * 32 * (8 * 32 * 8);
            for (int nt = 0; nt < 8; ++nt)
                nxt[nt] = *reinterpret_cast<const v8f*>(src + (long)nt * 32 * 8);
            if (ci + 2 < CC) {
                int cn3 = dir ? (CC - 3 - ci) : (ci + 2);
                const char* pf = (const char*)(gx +
                    (((long)dir * RT_TOT + (long)cn3 * 32 + bt) * 8) * 32 * 8) +
                    (long)lane * 256;
                __builtin_prefetch(pf, 0, 0);        // global_prefetch_b8
                __builtin_prefetch(pf + 128, 0, 0);  // cover full 8KB tile
            }
        }

        for (int nt = 0; nt < 8; ++nt)
            acc[nt] = wmma_bf16(hA, bW[nt], acc[nt]);

        for (int ht = 0; ht < 2; ++ht)
            for (int r = 0; r < 8; ++r) {
                float iv = sigf(acc[ht][r]);
                float fv = sigf(acc[2 + ht][r]);
                float gv = tanhf(acc[4 + ht][r]);
                float ov = sigf(acc[6 + ht][r]);
                float cv = fv * cacc[ht][r] + iv * gv;
                float hv = ov * tanhf(cv);
                cacc[ht][r] = cv;
                hcur[ht][r] = hv;
                int m = r + 8 * kh, n = ht * 16 + m15;
                __bf16 hb = tobf(hv);
                yout[((long)c * BB + b0 + m) * 64 + dir * HH + n] = hb;
                hbuf[m * 32 + n] = hb;
            }
        asm volatile("s_wait_dscnt 0" ::: "memory");   // LDS stores visible to reloads
        for (int j = 0; j < 16; ++j) hA[j] = hbuf[m15 * 32 + kf(kh, j)];
    }
    for (int ht = 0; ht < 2; ++ht)
        for (int r = 0; r < 8; ++r) {
            int m = r + 8 * kh, n = ht * 16 + m15;
            hout[((long)slice * BB + b0 + m) * HH + n] = hcur[ht][r];
            cio[((long)slice * BB + b0 + m) * HH + n] = cacc[ht][r];
        }
}

__global__ void fc_kernel(const __bf16* y1, const float* Wfc, const float* bfc,
                          float* xn, float* out, int t) {
    long i = (long)blockIdx.x * blockDim.x + threadIdx.x;
    if (i >= (long)BB * CC) return;
    int b = (int)(i / CC), c = (int)(i % CC);
    const __bf16* yr = y1 + ((long)c * BB + b) * 64;
    float s = bfc[0];
    for (int h = 0; h < HH; ++h) s += (float)yr[h] * Wfc[h];
    xn[(long)b * CC + c] = s;
    out[((long)b * PRED + t) * CC + c] = s;
}

// ---------------- host ----------------
extern "C" void kernel_launch(void* const* d_in, const int* in_sizes, int n_in,
                              void* d_out, int out_size, void* d_ws, size_t ws_size,
                              hipStream_t stream) {
    (void)in_sizes; (void)n_in; (void)out_size; (void)ws_size;
    const float* rain  = (const float*)d_in[0];
    const float* feat  = (const float*)d_in[1];
    const float* h0    = (const float*)d_in[2];
    const float* c0    = (const float*)d_in[3];
    const float* Wmlp  = (const float*)d_in[4];
    const float* bmlp  = (const float*)d_in[5];
    const float* Wihg  = (const float*)d_in[6];
    const float* Whhg  = (const float*)d_in[7];
    const float* bihg  = (const float*)d_in[8];
    const float* bhhg  = (const float*)d_in[9];
    const float* Wih0  = (const float*)d_in[10];
    const float* Whh0  = (const float*)d_in[11];
    const float* b0    = (const float*)d_in[12];
    const float* Wih1  = (const float*)d_in[13];
    const float* Whh1  = (const float*)d_in[14];
    const float* b1    = (const float*)d_in[15];
    const float* Wfc   = (const float*)d_in[16];
    const float* bfc   = (const float*)d_in[17];
    float* outp = (float*)d_out;

    char* ws = (char*)d_ws;
    size_t off = 0;
    auto carve = [&](size_t bytes) {
        char* p = ws + off;
        off = (off + bytes + 255) & ~(size_t)255;
        return p;
    };
    float*  xn    = (float*) carve((size_t)BB * CC * 4);
    float*  hnP   = (float*) carve((size_t)4 * BB * HH * 4);
    float*  hnG   = (float*) carve((size_t)4 * BB * HH * 4);
    float*  cn    = (float*) carve((size_t)4 * BB * HH * 4);
    float*  u     = (float*) carve((size_t)BB * 16 * 4);
    float*  xcat  = (float*) carve((size_t)CC * BB * DD * 4);
    __bf16* X14   = (__bf16*)carve((size_t)BB * KMLPP * 2);
    float*  gx    = (float*) carve((size_t)2 * RT_TOT * 8 * 32 * 8 * 4);
    __bf16* y0    = (__bf16*)carve((size_t)CC * BB * 64 * 2);
    __bf16* y1    = (__bf16*)carve((size_t)CC * BB * 64 * 2);
    __bf16* WmlpT = (__bf16*)carve((size_t)NMLPP * KMLPP * 2);
    __bf16* Wih0b = (__bf16*)carve((size_t)2 * 128 * 32 * 2);
    __bf16* Whh0b = (__bf16*)carve((size_t)2 * 128 * 32 * 2);
    __bf16* Wih1b = (__bf16*)carve((size_t)2 * 128 * 64 * 2);
    __bf16* Whh1b = (__bf16*)carve((size_t)2 * 128 * 32 * 2);

    {
        long tot = (long)NMLPP * KMLPP + 2 * 128 * 32 * 3 + 2 * 128 * 64;
        pack_weights_kernel<<<(unsigned)((tot + 255) / 256), 256, 0, stream>>>(
            Wmlp, Wih0, Whh0, Wih1, Whh1, WmlpT, Wih0b, Whh0b, Wih1b, Whh1b);
    }
    {
        long tot = (long)BB * CC + 2L * 4 * BB * HH;
        init_state_kernel<<<(unsigned)((tot + 255) / 256), 256, 0, stream>>>(
            rain, h0, c0, xn, hnP, cn);
    }

    for (int t = 0; t < PRED; ++t) {
        pack_x14_kernel<<<(unsigned)(((long)BB * KMLPP + 255) / 256), 256, 0, stream>>>(
            xn, feat, t, X14);
        mlp_gemm_kernel<<<96, 128, 0, stream>>>(X14, WmlpT, bmlp, xcat);
        build_xcat_kernel<<<(BB * CC + 255) / 256, 256, 0, stream>>>(xn, feat, t, xcat);
        mean_u_kernel<<<(BB * DD + 255) / 256, 256, 0, stream>>>(xcat, u);
        gru_kernel<<<(4 * BB * HH) / 256, 256, 0, stream>>>(u, hnP, Wihg, Whhg,
                                                            bihg, bhhg, hnG);
        gates_x_kernel<<<dim3(RT_TOT / 4, 2), 128, 0, stream>>>(xcat, nullptr, 0,
                                                                Wih0b, b0, gx);
        lstm_scan_kernel<<<dim3(BB / 16, 2), 32, 0, stream>>>(gx, Whh0b, y0, hnG,
                                                              hnP, cn, 0);
        gates_x_kernel<<<dim3(RT_TOT / 4, 2), 128, 0, stream>>>(nullptr, y0, 1,
                                                                Wih1b, b1, gx);
        lstm_scan_kernel<<<dim3(BB / 16, 2), 32, 0, stream>>>(gx, Whh1b, y1, hnG,
                                                              hnP, cn, 2);
        fc_kernel<<<(BB * CC + 255) / 256, 256, 0, stream>>>(y1, Wfc, bfc, xn, outp, t);
    }
}
